// DotProductAttn_22316650070307
// MI455X (gfx1250) — compile-verified, matrix operand order checked by reading, not run on hardware
//
#include <hip/hip_runtime.h>

// ---------------------------------------------------------------------------
// DotProductAttn for MI455X (gfx1250, wave32, WMMA)
//   scores[k,q] = keys[k]·queries[q]       (bf16 WMMA GEMM, f32 accum)
//   attn        = softmax over k (f32)
//   weighted    = attn^T · values          (bf16 WMMA GEMM, f32 accum)
// d_out = weighted (4096x1024 f32) ++ attn (4096x4096 f32)
// d_ws  = Kb, Qb, Vt, attnT bf16 staging (56 MB)
//
// GEMM: 256x128 block tile, 8 wave32s in 4x2 grid, 64x64 per wave
//       (16 v_wmma_f32_16x16x32_bf16 per wave per K-step, 4x fragment reuse),
//       double-buffered LDS (60 KB) -> global loads overlap WMMA, 1 barrier/step.
// ---------------------------------------------------------------------------

typedef __attribute__((ext_vector_type(16))) __bf16       v16bf;
typedef __attribute__((ext_vector_type(8)))  float        v8f;
typedef __attribute__((ext_vector_type(4)))  unsigned int u32x4;
typedef __attribute__((ext_vector_type(2)))  unsigned int u32x2;
typedef __attribute__((ext_vector_type(4)))  float        f32x4;
typedef unsigned short u16;

static __device__ __forceinline__ u16 f32_to_bf16(float x) {
  unsigned u = __builtin_bit_cast(unsigned, x);
  u += 0x7FFFu + ((u >> 16) & 1u);           // round-to-nearest-even
  return (u16)(u >> 16);
}

// fragment: 16 bf16 = 8 VGPRs, built from two 16-byte LDS loads
union Frag {
  v16bf bf;
  u32x4 q[2];
};

#define BM 256
#define BN 128
#define BK 32
#define LDSS 40   // LDS row stride in bf16 elems: 32 + 8 pad (bank de-conflict)

// C[m,n] = sum_k A[m,k] * B[n,k]   (both row-major [rows, K] bf16, C f32)
__global__ __launch_bounds__(256)
void gemm_bf16_nt(const u16* __restrict__ A, const u16* __restrict__ B,
                  float* __restrict__ C, int Kd, int lda, int ldb, int ldc) {
  __shared__ u16 sA[2][BM * LDSS];   // 2 * 20 KB
  __shared__ u16 sB[2][BN * LDSS];   // 2 * 10 KB   -> 60 KB total

  const int tid   = threadIdx.x;
  const int lane  = tid & 31;          // wave32
  const int wid   = tid >> 5;          // 8 waves / block
  const int waveM = wid >> 1;          // 0..3 -> 64 rows each
  const int waveN = wid & 1;           // 0..1 -> 64 cols each
  const int r     = lane & 15;         // row/col within 16x16 tile
  const int kb    = (lane >> 4) << 3;  // K chunk base: 0 or 8 (ISA A/B layout)

  const size_t rowA0 = (size_t)blockIdx.y * BM;
  const size_t rowB0 = (size_t)blockIdx.x * BN;

  const v8f vzero = {0.f, 0.f, 0.f, 0.f, 0.f, 0.f, 0.f, 0.f};
  v8f acc[4][4];
#pragma unroll
  for (int i = 0; i < 4; ++i)
#pragma unroll
    for (int j = 0; j < 4; ++j) acc[i][j] = vzero;

  // cooperative tile load mapping: thread handles column chunk (tid&3)*8,
  // rows (tid>>2) + i*64.  A: 4 chunks/thread, B: 2 chunks/thread.
  const int lrow = tid >> 2;
  const int lcol = (tid & 3) * 8;
  const u16* pA = A + (rowA0 + lrow) * lda + lcol;
  const u16* pB = B + (rowB0 + lrow) * ldb + lcol;
  const int S = Kd / BK;

  // ---- preload tile 0 into buffer 0 ----
  {
    u32x4 stA[4], stB[2];
#pragma unroll
    for (int i = 0; i < 4; ++i) stA[i] = *(const u32x4*)(pA + (size_t)i * 64 * lda);
#pragma unroll
    for (int i = 0; i < 2; ++i) stB[i] = *(const u32x4*)(pB + (size_t)i * 64 * ldb);
#pragma unroll
    for (int i = 0; i < 4; ++i)
      *(u32x4*)&sA[0][(lrow + i * 64) * LDSS + lcol] = stA[i];
#pragma unroll
    for (int i = 0; i < 2; ++i)
      *(u32x4*)&sB[0][(lrow + i * 64) * LDSS + lcol] = stB[i];
  }
  __syncthreads();

  for (int s = 0; s < S; ++s) {
    const int cur = s & 1, nxt = cur ^ 1;
    const bool has_next = (s + 1 < S);

    // ---- issue next tile's global loads (overlap with WMMA below) ----
    u32x4 stA[4], stB[2];
    if (has_next) {
      const int k1 = (s + 1) * BK;
#pragma unroll
      for (int i = 0; i < 4; ++i)
        stA[i] = *(const u32x4*)(pA + (size_t)i * 64 * lda + k1);
#pragma unroll
      for (int i = 0; i < 2; ++i)
        stB[i] = *(const u32x4*)(pB + (size_t)i * 64 * ldb + k1);
      if (s + 2 < S) {  // warm L2 for the tile after next (global_prefetch_b8)
        __builtin_prefetch(pA + (size_t)(s + 2) * BK, 0, 1);
        __builtin_prefetch(pB + (size_t)(s + 2) * BK, 0, 1);
      }
    }

    // ---- compute from current buffer ----
    Frag a[4], b[4];
#pragma unroll
    for (int tm = 0; tm < 4; ++tm) {
      const int base = (waveM * 64 + tm * 16 + r) * LDSS + kb;
      a[tm].q[0] = *(const u32x4*)&sA[cur][base];
      a[tm].q[1] = *(const u32x4*)&sA[cur][base + 16];
    }
#pragma unroll
    for (int tn = 0; tn < 4; ++tn) {
      const int base = (waveN * 64 + tn * 16 + r) * LDSS + kb;
      b[tn].q[0] = *(const u32x4*)&sB[cur][base];
      b[tn].q[1] = *(const u32x4*)&sB[cur][base + 16];
    }
#pragma unroll
    for (int tm = 0; tm < 4; ++tm)
#pragma unroll
      for (int tn = 0; tn < 4; ++tn)
        acc[tm][tn] = __builtin_amdgcn_wmma_f32_16x16x32_bf16(
            false, a[tm].bf, false, b[tn].bf, (short)0, acc[tm][tn], false, false);

    // ---- stage next tile into the other buffer (parity fenced by prev barrier)
    if (has_next) {
#pragma unroll
      for (int i = 0; i < 4; ++i)
        *(u32x4*)&sA[nxt][(lrow + i * 64) * LDSS + lcol] = stA[i];
#pragma unroll
      for (int i = 0; i < 2; ++i)
        *(u32x4*)&sB[nxt][(lrow + i * 64) * LDSS + lcol] = stB[i];
    }
    __syncthreads();
  }

  // epilogue: C/D layout -> lane holds N = lane&15, VGPR v holds M = v + 8*(lane>>4)
  const int rhi = lane >> 4;
#pragma unroll
  for (int tm = 0; tm < 4; ++tm)
#pragma unroll
    for (int tn = 0; tn < 4; ++tn) {
      const size_t gn = rowB0 + (size_t)(waveN * 64 + tn * 16 + r);
#pragma unroll
      for (int v = 0; v < 8; ++v) {
        const size_t gm = rowA0 + (size_t)(waveM * 64 + tm * 16 + v + 8 * rhi);
        C[gm * ldc + gn] = acc[tm][tn][v];
      }
    }
}

// numerically-stable online softmax over rows (axis k) for each column q
__global__ void softmax_cols(float* __restrict__ S, int Nk, int Nq) {
  int col = blockIdx.x * blockDim.x + threadIdx.x;
  if (col >= Nq) return;
  float m = -3.402823466e38f, s = 0.f;
  for (int k = 0; k < Nk; ++k) {
    float x  = S[(size_t)k * Nq + col];
    float mn = fmaxf(m, x);
    s = s * __expf(m - mn) + __expf(x - mn);
    m = mn;
  }
  float inv = 1.f / s;
  for (int k = 0; k < Nk; ++k) {
    size_t idx = (size_t)k * Nq + col;
    S[idx] = __expf(S[idx] - m) * inv;
  }
}

// dst[c, r] = bf16(src[r, c]);  src f32 [R, C] row-major, 32x32 LDS tiles
__global__ __launch_bounds__(256)
void transpose_f32_to_bf16(const float* __restrict__ src, u16* __restrict__ dst,
                           int R, int C) {
  __shared__ float t[32][33];
  int x0 = blockIdx.x * 32, y0 = blockIdx.y * 32;
#pragma unroll
  for (int j = 0; j < 4; ++j)
    t[threadIdx.y + 8 * j][threadIdx.x] =
        src[(size_t)(y0 + threadIdx.y + 8 * j) * C + x0 + threadIdx.x];
  __syncthreads();
#pragma unroll
  for (int j = 0; j < 4; ++j)
    dst[(size_t)(x0 + threadIdx.y + 8 * j) * R + y0 + threadIdx.x] =
        f32_to_bf16(t[threadIdx.x][threadIdx.y + 8 * j]);
}

// elementwise f32 -> bf16, 4 per thread
__global__ void cvt_f32_to_bf16_x4(const float* __restrict__ in,
                                   u16* __restrict__ out, int n4) {
  int i = blockIdx.x * blockDim.x + threadIdx.x;
  if (i >= n4) return;
  f32x4 v = ((const f32x4*)in)[i];
  u32x2 o;
  o.x = (unsigned)f32_to_bf16(v.x) | ((unsigned)f32_to_bf16(v.y) << 16);
  o.y = (unsigned)f32_to_bf16(v.z) | ((unsigned)f32_to_bf16(v.w) << 16);
  ((u32x2*)out)[i] = o;
}

extern "C" void kernel_launch(void* const* d_in, const int* in_sizes, int n_in,
                              void* d_out, int out_size, void* d_ws, size_t ws_size,
                              hipStream_t stream) {
  (void)in_sizes; (void)n_in; (void)out_size; (void)ws_size;
  constexpr int Nq = 4096, Nk = 4096, D = 1024;

  const float* Q = (const float*)d_in[0];
  const float* K = (const float*)d_in[1];
  const float* V = (const float*)d_in[2];

  float* weighted = (float*)d_out;                       // [Nq, D]
  float* attn     = (float*)d_out + (size_t)Nq * D;      // [Nk, Nq] (scores in-place)

  u16* Kb    = (u16*)d_ws;                               // [Nk, D]  bf16
  u16* Qb    = Kb + (size_t)Nk * D;                      // [Nq, D]  bf16
  u16* Vt    = Qb + (size_t)Nq * D;                      // [D,  Nk] bf16
  u16* attnT = Vt + (size_t)D * Nk;                      // [Nq, Nk] bf16

  // 1) cast K, Q to bf16; transpose+cast V -> Vt
  cvt_f32_to_bf16_x4<<<(Nk * D / 4 + 255) / 256, 256, 0, stream>>>(K, Kb, Nk * D / 4);
  cvt_f32_to_bf16_x4<<<(Nq * D / 4 + 255) / 256, 256, 0, stream>>>(Q, Qb, Nq * D / 4);
  transpose_f32_to_bf16<<<dim3(D / 32, Nk / 32), dim3(32, 8), 0, stream>>>(V, Vt, Nk, D);

  // 2) scores[k,q] = Kb · Qb^T   (M=Nk, N=Nq, Kd=D), written into attn region
  gemm_bf16_nt<<<dim3(Nq / BN, Nk / BM), 256, 0, stream>>>(Kb, Qb, attn, D, D, D, Nq);

  // 3) softmax over k for each column q (in place, f32)
  softmax_cols<<<Nq / 256, 256, 0, stream>>>(attn, Nk, Nq);

  // 4) attnT[q,k] = bf16(attn[k,q])
  transpose_f32_to_bf16<<<dim3(Nq / 32, Nk / 32), dim3(32, 8), 0, stream>>>(attn, attnT, Nk, Nq);

  // 5) weighted[q,e] = attnT · Vt^T  (M=Nq, N=D, Kd=Nk)
  gemm_bf16_nt<<<dim3(D / BN, Nq / BM), 256, 0, stream>>>(attnT, Vt, weighted, Nk, Nk, Nk, D);
}
